// Net_13116830122564
// MI455X (gfx1250) — compile-verified, matrix operand order checked by reading, not run on hardware
//
#include <hip/hip_runtime.h>

// ---------------------------------------------------------------------------
// CDNA5 (gfx1250) WMMA implementation of the voxel-PointNet + conv pyramid +
// object-PointNet + per-cell classifier network.
//
// All heavy GEMMs use v_wmma_f32_16x16x32_f16 (wave32). Each wave owns a
// 16(M) x 64(N) block (4 accumulators) so one A fragment feeds 4 WMMAs;
// fused LDS kernels hoist all A fragments of a layer into registers.
// Workspace usage ~87 MB.
// ---------------------------------------------------------------------------

typedef __attribute__((ext_vector_type(16))) _Float16 v16h;
typedef __attribute__((ext_vector_type(8)))  float    v8f;

#define NB   32
#define PSC  8192
#define POB  2048
#define CUT  20

// k-index pattern for 16-bit A/B fragments (ISA 7.12.2, 16-bit 16x32 A):
// lanes 0-15 hold rows 0-15 with K = {v*2, v*2+1} for VGPR v in 0..3 and
// K = 16 + ... for v in 4..7; lanes 16-31 add +8.
__device__ __forceinline__ int frag_k(int i, int lane) {
  return ((i >> 3) << 4) | (((lane >> 4) & 1) << 3) | (((i >> 1) & 3) << 1) | (i & 1);
}

// Load a 16x32 fragment from a row-major [rows][ld] f16 matrix.
// Row for lane l is row0 + (l & 15); works for A ([M][K]) and for B when
// weights are stored N-major ([N][K]) — k-pairs are contiguous in memory.
__device__ __forceinline__ v16h load_frag_nk(const _Float16* __restrict__ p,
                                             int ld, int row0, int k0, int lane) {
  const _Float16* q = p + (size_t)(row0 + (lane & 15)) * ld + k0;
  v16h f;
#pragma unroll
  for (int i = 0; i < 16; ++i) f[i] = q[frag_k(i, lane)];
  return f;
}

__device__ __forceinline__ v8f wmma_f16(v16h a, v16h b, v8f c) {
  return __builtin_amdgcn_wmma_f32_16x16x32_f16(false, a, false, b, (short)0, c,
                                                false, false);
}

// ---------------------------------------------------------------------------
// Utility kernels
// ---------------------------------------------------------------------------
__global__ void k_zero_f32(float* __restrict__ p, int n) {
  int i = blockIdx.x * blockDim.x + threadIdx.x;
  if (i < n) p[i] = 0.f;
}

__global__ void k_cvt_f16(const float* __restrict__ src, _Float16* __restrict__ dst, int n) {
  int i = blockIdx.x * blockDim.x + threadIdx.x;
  if (i < n) dst[i] = (_Float16)src[i];
}

// cls_w1: [1024][1538] f32 -> [1024][1568] f16 (zero-padded K)
__global__ void k_prep_cls_w1(const float* __restrict__ src, _Float16* __restrict__ dst) {
  int i = blockIdx.x * blockDim.x + threadIdx.x;
  if (i >= 1024 * 1568) return;
  int o = i / 1568, k = i - o * 1568;
  dst[i] = (_Float16)(k < 1538 ? src[o * 1538 + k] : 0.f);
}

// convt_w: [512][1024][2][2] f32 -> Wt[tap=ky*2+kx][512][1024] f16
__global__ void k_prep_convt(const float* __restrict__ src, _Float16* __restrict__ dst) {
  int i = blockIdx.x * blockDim.x + threadIdx.x;
  if (i >= 4 * 512 * 1024) return;
  int tap = i >> 19;
  int rem = i & ((512 * 1024) - 1);
  int o = rem >> 10, ci = rem & 1023;
  int ky = tap >> 1, kx = tap & 1;
  dst[i] = (_Float16)src[((o * 1024 + ci) * 2 + ky) * 2 + kx];
}

// Per-batch mean of object points -> mean[32][2]
__global__ void k_ocmean(const float* __restrict__ oc, float* __restrict__ mean) {
  __shared__ float sx[256], sy[256];
  int n = blockIdx.x, t = threadIdx.x;
  float ax = 0.f, ay = 0.f;
  for (int p = t; p < POB; p += 256) {
    ax += oc[(n * POB + p) * 2 + 0];
    ay += oc[(n * POB + p) * 2 + 1];
  }
  sx[t] = ax; sy[t] = ay;
  __syncthreads();
  for (int s = 128; s > 0; s >>= 1) {
    if (t < s) { sx[t] += sx[t + s]; sy[t] += sy[t + s]; }
    __syncthreads();
  }
  if (t == 0) {
    mean[n * 2 + 0] = sx[0] * (1.f / POB);
    mean[n * 2 + 1] = sy[0] * (1.f / POB);
  }
}

// 2x2 max-pool on [NC][H][W] f16
__global__ void k_pool(const _Float16* __restrict__ src, _Float16* __restrict__ dst,
                       int NC, int H, int W) {
  int Ho = H >> 1, Wo = W >> 1;
  int idx = blockIdx.x * blockDim.x + threadIdx.x;
  if (idx >= NC * Ho * Wo) return;
  int r = idx / (Ho * Wo), rem = idx - r * (Ho * Wo);
  int y = rem / Wo, x = rem - y * Wo;
  const _Float16* s = src + ((size_t)r * H + 2 * y) * W + 2 * x;
  float a = (float)s[0], b = (float)s[1], c = (float)s[W], d = (float)s[W + 1];
  dst[idx] = (_Float16)fmaxf(fmaxf(a, b), fmaxf(c, d));
}

// ---------------------------------------------------------------------------
// Fused voxel PointNet: 16 points per block, layer1 (2->128) scalar into LDS,
// layer2 (128->256) via WMMA with hoisted A fragments, epilogue = ReLU +
// BRANCHLESS atomic max-scatter (invalid points go to a dummy voxel slab,
// exactly the reference's dummy-bin trick; ReLU >= 0 makes uint-bit atomicMax
// order-equivalent to float max with 0-init).
// ---------------------------------------------------------------------------
__global__ void k_voxel(const float* __restrict__ sc,
                        const float* __restrict__ w1, const float* __restrict__ b1,
                        const _Float16* __restrict__ W2, const float* __restrict__ b2,
                        float* __restrict__ vox) {
  __shared__ _Float16 A[16 * 128];
  __shared__ float pcx[16], pcy[16];
  __shared__ int lbase[16];
  int t = threadIdx.x;
  int m0 = blockIdx.x * 16;
  if (t < 16) {
    int g = m0 + t;
    int n = g >> 13;                       // /PSC
    float x = sc[2 * g], y = sc[2 * g + 1];
    int bi = (int)ceilf((x + 0.5f) * (float)CUT) - 1; bi = min(max(bi, 0), CUT - 1);
    int bj = (int)ceilf((y + 0.5f) * (float)CUT) - 1; bj = min(max(bj, 0), CUT - 1);
    bool valid = (x > -0.5f) && (x <= 0.5f) && (y > -0.5f) && (y <= 0.5f);
    float ci = -0.5f + (bi + 0.5f) * (1.0f / CUT);
    float cj = -0.5f + (bj + 0.5f) * (1.0f / CUT);
    pcx[t] = x - ci; pcy[t] = y - cj;
    // invalid -> dummy slab at batch index NB (zeroed, never read back)
    lbase[t] = valid ? (n * (256 * CUT * CUT) + bi * CUT + bj)
                     : (NB * (256 * CUT * CUT));
  }
  __syncthreads();
  for (int idx = t; idx < 16 * 128; idx += blockDim.x) {
    int p = idx >> 7, c = idx & 127;
    float h = fmaf(w1[2 * c], pcx[p], fmaf(w1[2 * c + 1], pcy[p], b1[c]));
    A[idx] = (_Float16)fmaxf(h, 0.f);
  }
  __syncthreads();
  int wave = t >> 5, lane = t & 31;
  int crow = (lane >> 4) << 3, ccol = lane & 15;
  // hoist the 4 A fragments (K=128) once; reuse across all 4 N tiles
  v16h af[4];
#pragma unroll
  for (int kk = 0; kk < 4; ++kk) af[kk] = load_frag_nk(A, 128, 0, kk * 32, lane);
  int base0 = lbase[crow + 0], base1 = lbase[crow + 1];
  int base2 = lbase[crow + 2], base3 = lbase[crow + 3];
  int base4 = lbase[crow + 4], base5 = lbase[crow + 5];
  int base6 = lbase[crow + 6], base7 = lbase[crow + 7];
#pragma unroll
  for (int tile = 0; tile < 4; ++tile) {        // 4 waves x 4 tiles = 256 cols
    int n0 = wave * 64 + tile * 16;
    v8f acc = {};
#pragma unroll
    for (int kk = 0; kk < 4; ++kk) {
      v16h b = load_frag_nk(W2, 128, n0, kk * 32, lane);
      acc = wmma_f16(af[kk], b, acc);
    }
    int c = n0 + ccol;
    float bv = b2[c];
    int coff = c * (CUT * CUT);
    unsigned int* vp = (unsigned int*)vox;
    atomicMax(vp + base0 + coff, __float_as_uint(fmaxf(acc[0] + bv, 0.f)));
    atomicMax(vp + base1 + coff, __float_as_uint(fmaxf(acc[1] + bv, 0.f)));
    atomicMax(vp + base2 + coff, __float_as_uint(fmaxf(acc[2] + bv, 0.f)));
    atomicMax(vp + base3 + coff, __float_as_uint(fmaxf(acc[3] + bv, 0.f)));
    atomicMax(vp + base4 + coff, __float_as_uint(fmaxf(acc[4] + bv, 0.f)));
    atomicMax(vp + base5 + coff, __float_as_uint(fmaxf(acc[5] + bv, 0.f)));
    atomicMax(vp + base6 + coff, __float_as_uint(fmaxf(acc[6] + bv, 0.f)));
    atomicMax(vp + base7 + coff, __float_as_uint(fmaxf(acc[7] + bv, 0.f)));
  }
}

// ---------------------------------------------------------------------------
// Implicit-GEMM 3x3 same-conv + ReLU: src [nb][Cin][H][W] f16 -> dst f16.
// One wave = 16(M) x 64(Cout) block: the expensive bounds-checked im2col A
// gather is done once per k-step and feeds 4 WMMAs. K = Cin*9 (mult of 32).
// ---------------------------------------------------------------------------
__global__ void k_conv3x3(const _Float16* __restrict__ src, const _Float16* __restrict__ Wf,
                          const float* __restrict__ bias, _Float16* __restrict__ dst,
                          int Cin, int H, int W, int Cout, int M) {
  int wid = (blockIdx.x * blockDim.x + threadIdx.x) >> 5;
  int lane = threadIdx.x & 31;
  int ngrp = Cout >> 6;                       // 64-col groups
  if (wid >= (M >> 4) * ngrp) return;
  int mt = wid / ngrp, nt = wid - mt * ngrp;
  int m0 = mt << 4, n0 = nt << 6;
  int hw = H * W;
  int row = m0 + (lane & 15);
  int nb = row / hw, rem = row - nb * hw;
  int y = rem / W, x = rem - y * W;
  const _Float16* sbase = src + (size_t)nb * Cin * hw;
  int K = Cin * 9;
  v8f acc[4] = {};
  for (int k0 = 0; k0 < K; k0 += 32) {
    v16h a;
#pragma unroll
    for (int i = 0; i < 16; ++i) {
      int k = k0 + frag_k(i, lane);
      int c = k / 9, r9 = k - c * 9;
      int ky = r9 / 3, kx = r9 - ky * 3;
      int iy = y + ky - 1, ix = x + kx - 1;
      _Float16 v = (_Float16)0.f;
      if ((unsigned)iy < (unsigned)H && (unsigned)ix < (unsigned)W)
        v = sbase[(size_t)c * hw + iy * W + ix];
      a[i] = v;
    }
#pragma unroll
    for (int j = 0; j < 4; ++j) {
      v16h b = load_frag_nk(Wf, K, n0 + 16 * j, k0, lane);
      acc[j] = wmma_f16(a, b, acc[j]);
    }
  }
  int crow = (lane >> 4) << 3, ccol = lane & 15;
#pragma unroll
  for (int j = 0; j < 4; ++j) {
    int co = n0 + 16 * j + ccol;
    float bv = bias[co];
#pragma unroll
    for (int r = 0; r < 8; ++r) {
      int m = m0 + r + crow;
      int nb2 = m / hw, rem2 = m - nb2 * hw;
      dst[(size_t)(nb2 * Cout + co) * hw + rem2] = (_Float16)fmaxf(acc[j][r] + bv, 0.f);
    }
  }
}

// ---------------------------------------------------------------------------
// Conv-transpose 2x2 stride 2 (non-overlapping): one GEMM per output parity.
// out(2i+py, 2j+px) = sum_c f2p[n][c][i][j] * Wt[tap][co][c] + b   (no ReLU)
// One wave = 16 x 64 block.
// ---------------------------------------------------------------------------
__global__ void k_convt(const _Float16* __restrict__ f2p, const _Float16* __restrict__ Wt,
                        const float* __restrict__ bias, _Float16* __restrict__ f3,
                        int py, int px) {
  int wid = (blockIdx.x * blockDim.x + threadIdx.x) >> 5;
  int lane = threadIdx.x & 31;
  if (wid >= 50 * 8) return;                  // M=800 -> 50 tiles, N=512 -> 8 groups
  int mt = wid >> 3, nt = wid & 7;
  int m0 = mt << 4, n0 = nt << 6;
  int row = m0 + (lane & 15);
  int nb = row / 25, rem = row - nb * 25;
  int i = rem / 5, j = rem - i * 5;
  const _Float16* sbase = f2p + (size_t)nb * 1024 * 25 + i * 5 + j;
  v8f acc[4] = {};
  for (int k0 = 0; k0 < 1024; k0 += 32) {
    v16h a;
#pragma unroll
    for (int ii = 0; ii < 16; ++ii) a[ii] = sbase[(k0 + frag_k(ii, lane)) * 25];
#pragma unroll
    for (int jj = 0; jj < 4; ++jj) {
      v16h b = load_frag_nk(Wt, 1024, n0 + 16 * jj, k0, lane);
      acc[jj] = wmma_f16(a, b, acc[jj]);
    }
  }
  int crow = (lane >> 4) << 3, ccol = lane & 15;
#pragma unroll
  for (int jj = 0; jj < 4; ++jj) {
    int co = n0 + 16 * jj + ccol;
    float bv = bias[co];
#pragma unroll
    for (int r = 0; r < 8; ++r) {
      int m = m0 + r + crow;
      int nb2 = m / 25, rem2 = m - nb2 * 25;
      int i2 = rem2 / 5, j2 = rem2 - i2 * 5;
      int oy = 2 * i2 + py, ox = 2 * j2 + px;
      f3[((size_t)(nb2 * 512 + co) * 10 + oy) * 10 + ox] = (_Float16)(acc[jj][r] + bv);
    }
  }
}

// ---------------------------------------------------------------------------
// Fused object set-abstraction: 16 points/block through 2->64->128->256->512
// with LDS activations, per-layer hoisted A fragments, and a fused global
// max-reduce into obj[32][512].
// ---------------------------------------------------------------------------
__global__ void k_sa(const float* __restrict__ oc, const float* __restrict__ mean,
                     const float* __restrict__ w1, const float* __restrict__ b1,
                     const _Float16* __restrict__ W2, const float* __restrict__ b2,
                     const _Float16* __restrict__ W3, const float* __restrict__ b3,
                     const _Float16* __restrict__ W4, const float* __restrict__ b4,
                     float* __restrict__ obj) {
  __shared__ _Float16 A1[16 * 64], A2[16 * 128], A3[16 * 256];
  __shared__ float cx[16], cy[16];
  int t = threadIdx.x;
  int m0 = blockIdx.x * 16;
  int n = m0 >> 11;                                  // /POB, same for all 16 pts
  if (t < 16) {
    int g = m0 + t;
    cx[t] = oc[2 * g]     - mean[n * 2];
    cy[t] = oc[2 * g + 1] - mean[n * 2 + 1];
  }
  __syncthreads();
  for (int idx = t; idx < 16 * 64; idx += blockDim.x) {
    int p = idx >> 6, c = idx & 63;
    float h = fmaf(w1[2 * c], cx[p], fmaf(w1[2 * c + 1], cy[p], b1[c]));
    A1[idx] = (_Float16)fmaxf(h, 0.f);
  }
  __syncthreads();
  int wave = t >> 5, lane = t & 31;
  int crow = (lane >> 4) << 3, ccol = lane & 15;
  // ---- layer2: 64 -> 128 (2 A frags, 2 tiles/wave) ----
  {
    v16h af[2];
#pragma unroll
    for (int kk = 0; kk < 2; ++kk) af[kk] = load_frag_nk(A1, 64, 0, kk * 32, lane);
#pragma unroll
    for (int tile = 0; tile < 2; ++tile) {
      int n0 = wave * 32 + tile * 16;
      v8f acc = {};
#pragma unroll
      for (int kk = 0; kk < 2; ++kk) {
        v16h b = load_frag_nk(W2, 64, n0, kk * 32, lane);
        acc = wmma_f16(af[kk], b, acc);
      }
      int c = n0 + ccol; float bv = b2[c];
#pragma unroll
      for (int r = 0; r < 8; ++r)
        A2[(r + crow) * 128 + c] = (_Float16)fmaxf(acc[r] + bv, 0.f);
    }
  }
  __syncthreads();
  // ---- layer3: 128 -> 256 (4 A frags, 4 tiles/wave) ----
  {
    v16h af[4];
#pragma unroll
    for (int kk = 0; kk < 4; ++kk) af[kk] = load_frag_nk(A2, 128, 0, kk * 32, lane);
#pragma unroll
    for (int tile = 0; tile < 4; ++tile) {
      int n0 = wave * 64 + tile * 16;
      v8f acc = {};
#pragma unroll
      for (int kk = 0; kk < 4; ++kk) {
        v16h b = load_frag_nk(W3, 128, n0, kk * 32, lane);
        acc = wmma_f16(af[kk], b, acc);
      }
      int c = n0 + ccol; float bv = b3[c];
#pragma unroll
      for (int r = 0; r < 8; ++r)
        A3[(r + crow) * 256 + c] = (_Float16)fmaxf(acc[r] + bv, 0.f);
    }
  }
  __syncthreads();
  // ---- layer4: 256 -> 512 (8 A frags, 8 tiles/wave), fused max over points ----
  {
    v16h af[8];
#pragma unroll
    for (int kk = 0; kk < 8; ++kk) af[kk] = load_frag_nk(A3, 256, 0, kk * 32, lane);
#pragma unroll
    for (int tile = 0; tile < 8; ++tile) {
      int n0 = wave * 128 + tile * 16;
      v8f acc = {};
#pragma unroll
      for (int kk = 0; kk < 8; ++kk) {
        v16h b = load_frag_nk(W4, 256, n0, kk * 32, lane);
        acc = wmma_f16(af[kk], b, acc);
      }
      int c = n0 + ccol; float bv = b4[c];
      float mx = 0.f;
#pragma unroll
      for (int r = 0; r < 8; ++r) mx = fmaxf(mx, fmaxf(acc[r] + bv, 0.f));
      atomicMax((unsigned int*)&obj[n * 512 + c], __float_as_uint(mx));
    }
  }
}

// ---------------------------------------------------------------------------
// Assemble classifier input A [3200][1568] f16 (scene | obj | rel | pad)
// ---------------------------------------------------------------------------
__global__ void k_clsin(const _Float16* __restrict__ f1p, const _Float16* __restrict__ f3,
                        const float* __restrict__ obj, const float* __restrict__ pos,
                        _Float16* __restrict__ Acls) {
  const int Ktot = 1568;
  int idx = blockIdx.x * blockDim.x + threadIdx.x;
  if (idx >= 3200 * Ktot) return;
  int m = idx / Ktot, k = idx - m * Ktot;
  int n = m / 100, pp = m - n * 100;
  int y = pp / 10, x = pp - y * 10;
  float v;
  if (k < 512)        v = (float)f1p[(size_t)(n * 512 + k) * 100 + pp];
  else if (k < 1024)  v = (float)f3[(size_t)(n * 512 + (k - 512)) * 100 + pp];
  else if (k < 1536)  v = obj[n * 512 + (k - 1024)];
  else if (k == 1536) v = pos[n * 2]     - (-0.5f + (y + 0.5f) * 0.1f);
  else if (k == 1537) v = pos[n * 2 + 1] - (-0.5f + (x + 0.5f) * 0.1f);
  else                v = 0.f;
  Acls[idx] = (_Float16)v;
}

// Generic WMMA GEMM + bias (+ReLU): C[M][N] = act(A[M][K] @ W[N][K]^T + b)
// One wave = 16 x 64 block (4 accumulators share each A fragment).
__global__ void k_gemm_relu(const _Float16* __restrict__ A, const _Float16* __restrict__ W,
                            const float* __restrict__ bias, _Float16* __restrict__ C,
                            int M, int N, int K, int relu) {
  int wid = (blockIdx.x * blockDim.x + threadIdx.x) >> 5;
  int lane = threadIdx.x & 31;
  int ngrp = N >> 6;
  if (wid >= (M >> 4) * ngrp) return;
  int mt = wid / ngrp, nt = wid - mt * ngrp;
  int m0 = mt << 4, n0 = nt << 6;
  v8f acc[4] = {};
  for (int k0 = 0; k0 < K; k0 += 32) {
    v16h a = load_frag_nk(A, K, m0, k0, lane);
#pragma unroll
    for (int j = 0; j < 4; ++j) {
      v16h b = load_frag_nk(W, K, n0 + 16 * j, k0, lane);
      acc[j] = wmma_f16(a, b, acc[j]);
    }
  }
  int crow = (lane >> 4) << 3, ccol = lane & 15;
#pragma unroll
  for (int j = 0; j < 4; ++j) {
    int co = n0 + 16 * j + ccol;
    float bv = bias[co];
#pragma unroll
    for (int r = 0; r < 8; ++r) {
      int m = m0 + r + crow;
      float v = acc[j][r] + bv;
      if (relu) v = fmaxf(v, 0.f);
      C[(size_t)m * N + co] = (_Float16)v;
    }
  }
}

// Final 256 -> 1 projection (f32 accumulate, f32 output)
__global__ void k_cls3(const _Float16* __restrict__ h2, const float* __restrict__ w3,
                       const float* __restrict__ b3, float* __restrict__ out) {
  int m = blockIdx.x * blockDim.x + threadIdx.x;
  if (m >= 3200) return;
  float s = b3[0];
  for (int k = 0; k < 256; ++k) s += (float)h2[(size_t)m * 256 + k] * w3[k];
  out[m] = s;
}

// ---------------------------------------------------------------------------
// Host-side orchestration
// ---------------------------------------------------------------------------
static inline int blks(long long n, int tpb) { return (int)((n + tpb - 1) / tpb); }

extern "C" void kernel_launch(void* const* d_in, const int* in_sizes, int n_in,
                              void* d_out, int out_size, void* d_ws, size_t ws_size,
                              hipStream_t stream) {
  (void)in_sizes; (void)n_in; (void)out_size; (void)ws_size;
  const float* sc      = (const float*)d_in[0];
  const float* oc      = (const float*)d_in[1];
  const float* pos     = (const float*)d_in[2];
  const float* mlp_w1  = (const float*)d_in[3];
  const float* mlp_b1  = (const float*)d_in[4];
  const float* mlp_w2  = (const float*)d_in[5];
  const float* mlp_b2  = (const float*)d_in[6];
  const float* conv1_w = (const float*)d_in[7];
  const float* conv1_b = (const float*)d_in[8];
  const float* conv2_w = (const float*)d_in[9];
  const float* conv2_b = (const float*)d_in[10];
  const float* convt_w = (const float*)d_in[11];
  const float* convt_b = (const float*)d_in[12];
  const float* sa_w1   = (const float*)d_in[13];
  const float* sa_b1   = (const float*)d_in[14];
  const float* sa_w2   = (const float*)d_in[15];
  const float* sa_b2   = (const float*)d_in[16];
  const float* sa_w3   = (const float*)d_in[17];
  const float* sa_b3   = (const float*)d_in[18];
  const float* sa_w4   = (const float*)d_in[19];
  const float* sa_b4   = (const float*)d_in[20];
  const float* cls_w1  = (const float*)d_in[21];
  const float* cls_b1  = (const float*)d_in[22];
  const float* cls_w2  = (const float*)d_in[23];
  const float* cls_b2  = (const float*)d_in[24];
  const float* cls_w3  = (const float*)d_in[25];
  const float* cls_b3  = (const float*)d_in[26];
  float* out = (float*)d_out;

  // ---- workspace layout (bytes, 256-aligned; total ~87 MB) ----
  char* ws = (char*)d_ws;
  size_t off = 0;
  auto alloc = [&](size_t bytes) { size_t o = off; off = (off + bytes + 255) & ~(size_t)255; return o; };
  _Float16* Wmlp2 = (_Float16*)(ws + alloc(256 * 128 * 2));
  _Float16* Wc1   = (_Float16*)(ws + alloc((size_t)512 * 2304 * 2));
  _Float16* Wc2   = (_Float16*)(ws + alloc((size_t)1024 * 4608 * 2));
  _Float16* Wct   = (_Float16*)(ws + alloc((size_t)4 * 512 * 1024 * 2));
  _Float16* Wsa2  = (_Float16*)(ws + alloc(128 * 64 * 2));
  _Float16* Wsa3  = (_Float16*)(ws + alloc(256 * 128 * 2));
  _Float16* Wsa4  = (_Float16*)(ws + alloc(512 * 256 * 2));
  _Float16* Wcls1 = (_Float16*)(ws + alloc((size_t)1024 * 1568 * 2));
  _Float16* Wcls2 = (_Float16*)(ws + alloc((size_t)256 * 1024 * 2));
  // voxel grid + one dummy batch-slab for invalid points (branchless scatter)
  float*    vox   = (float*)   (ws + alloc((size_t)(NB + 1) * 256 * 400 * 4));
  _Float16* voxh  = (_Float16*)(ws + alloc((size_t)NB * 256 * 400 * 2));
  _Float16* f1    = (_Float16*)(ws + alloc((size_t)NB * 512 * 400 * 2));
  _Float16* f1p   = (_Float16*)(ws + alloc((size_t)NB * 512 * 100 * 2));
  _Float16* f2    = (_Float16*)(ws + alloc((size_t)NB * 1024 * 100 * 2));
  _Float16* f2p   = (_Float16*)(ws + alloc((size_t)NB * 1024 * 25 * 2));
  _Float16* f3    = (_Float16*)(ws + alloc((size_t)NB * 512 * 100 * 2));
  float*    obj   = (float*)   (ws + alloc(NB * 512 * 4));
  float*    mean  = (float*)   (ws + alloc(NB * 2 * 4));
  _Float16* Acls  = (_Float16*)(ws + alloc((size_t)3200 * 1568 * 2));
  _Float16* h1    = (_Float16*)(ws + alloc((size_t)3200 * 1024 * 2));
  _Float16* h2    = (_Float16*)(ws + alloc((size_t)3200 * 256 * 2));

  // ---- weight prep (f32 -> f16, required layouts) ----
  k_cvt_f16<<<blks(256 * 128, 256), 256, 0, stream>>>(mlp_w2, Wmlp2, 256 * 128);
  k_cvt_f16<<<blks(512 * 2304, 256), 256, 0, stream>>>(conv1_w, Wc1, 512 * 2304);
  k_cvt_f16<<<blks(1024 * 4608, 256), 256, 0, stream>>>(conv2_w, Wc2, 1024 * 4608);
  k_prep_convt<<<blks(4 * 512 * 1024, 256), 256, 0, stream>>>(convt_w, Wct);
  k_cvt_f16<<<blks(128 * 64, 256), 256, 0, stream>>>(sa_w2, Wsa2, 128 * 64);
  k_cvt_f16<<<blks(256 * 128, 256), 256, 0, stream>>>(sa_w3, Wsa3, 256 * 128);
  k_cvt_f16<<<blks(512 * 256, 256), 256, 0, stream>>>(sa_w4, Wsa4, 512 * 256);
  k_prep_cls_w1<<<blks(1024 * 1568, 256), 256, 0, stream>>>(cls_w1, Wcls1);
  k_cvt_f16<<<blks(256 * 1024, 256), 256, 0, stream>>>(cls_w2, Wcls2, 256 * 1024);

  // ---- zero-init scatter targets (every call; deterministic) ----
  k_zero_f32<<<blks((NB + 1) * 256 * 400, 256), 256, 0, stream>>>(vox, (NB + 1) * 256 * 400);
  k_zero_f32<<<blks(NB * 512, 256), 256, 0, stream>>>(obj, NB * 512);

  // ---- voxel PointNet + scatter-max ----
  k_voxel<<<(NB * PSC) / 16, 128, 0, stream>>>(sc, mlp_w1, mlp_b1, Wmlp2, mlp_b2, vox);
  k_cvt_f16<<<blks(NB * 256 * 400, 256), 256, 0, stream>>>(vox, voxh, NB * 256 * 400);

  // ---- conv pyramid (implicit-GEMM WMMA, 16x64 blocks/wave) ----
  {
    int M = NB * 400;                                        // 12800
    int waves = (M / 16) * (512 / 64);                       // 6400
    k_conv3x3<<<blks((long long)waves * 32, 256), 256, 0, stream>>>(
        voxh, Wc1, conv1_b, f1, 256, 20, 20, 512, M);
  }
  k_pool<<<blks((long long)NB * 512 * 100, 256), 256, 0, stream>>>(f1, f1p, NB * 512, 20, 20);
  {
    int M = NB * 100;                                        // 3200
    int waves = (M / 16) * (1024 / 64);                      // 3200
    k_conv3x3<<<blks((long long)waves * 32, 256), 256, 0, stream>>>(
        f1p, Wc2, conv2_b, f2, 512, 10, 10, 1024, M);
  }
  k_pool<<<blks((long long)NB * 1024 * 25, 256), 256, 0, stream>>>(f2, f2p, NB * 1024, 10, 10);

  // conv_transpose 2x2 s2 (VALID, no kernel flip in lax.conv_transpose =>
  // output parity (py,px) uses tap (1-py, 1-px))
  for (int py = 0; py < 2; ++py)
    for (int px = 0; px < 2; ++px) {
      int tap = (1 - py) * 2 + (1 - px);
      k_convt<<<blks((long long)400 * 32, 256), 256, 0, stream>>>(
          f2p, Wct + (size_t)tap * 512 * 1024, convt_b, f3, py, px);
    }

  // ---- object PointNet (fused 4-layer WMMA chain + global max) ----
  k_ocmean<<<NB, 256, 0, stream>>>(oc, mean);
  k_sa<<<(NB * POB) / 16, 128, 0, stream>>>(oc, mean, sa_w1, sa_b1,
                                            Wsa2, sa_b2, Wsa3, sa_b3, Wsa4, sa_b4, obj);

  // ---- per-cell classifier ----
  k_clsin<<<blks((long long)3200 * 1568, 256), 256, 0, stream>>>(f1p, f3, obj, pos, Acls);
  {
    int waves = (3200 / 16) * (1024 / 64);                   // 3200
    k_gemm_relu<<<blks((long long)waves * 32, 256), 256, 0, stream>>>(
        Acls, Wcls1, cls_b1, h1, 3200, 1024, 1568, 1);
  }
  {
    int waves = (3200 / 16) * (256 / 64);                    // 800
    k_gemm_relu<<<blks((long long)waves * 32, 256), 256, 0, stream>>>(
        h1, Wcls2, cls_b2, h2, 3200, 256, 1024, 1);
  }
  k_cls3<<<blks(3200, 256), 256, 0, stream>>>(h2, cls_w3, cls_b3, out);
}
